// ViT_11879879544436
// MI455X (gfx1250) — matrix operand order repaired and checked by reading, not compiled, not run on hardware
//
#include <hip/hip_runtime.h>

#define BATCH    512
#define DMODEL   256
#define LAYERS   6
#define NHEADS   4
#define NTOKENS  65
#define NCLASSES 100
#define MROWS    (BATCH * NTOKENS)   // 33280, divisible by 16

typedef __attribute__((ext_vector_type(16))) _Float16 v16h;
typedef __attribute__((ext_vector_type(8)))  _Float16 v8h;
typedef __attribute__((ext_vector_type(8)))  float    v8f;

union HF16 { v16h v; v8h h[2]; };

__device__ __forceinline__ float wave_sum(float v) {
#pragma unroll
  for (int o = 16; o > 0; o >>= 1) v += __shfl_xor(v, o, 32);
  return v;
}
__device__ __forceinline__ float wave_max(float v) {
#pragma unroll
  for (int o = 16; o > 0; o >>= 1) v = fmaxf(v, __shfl_xor(v, o, 32));
  return v;
}

// ---------------------------------------------------------------------------
// Patch embedding: conv 4x4/s4 + cls token + pos embed -> x f32 [M, 256]
// ---------------------------------------------------------------------------
__global__ __launch_bounds__(256) void patch_embed_kernel(
    const float* __restrict__ inp, const float* __restrict__ pw,
    const float* __restrict__ pb, const float* __restrict__ cls,
    const float* __restrict__ pos, float* __restrict__ x) {
  int idx = blockIdx.x * 256 + threadIdx.x;
  int d = idx & 255;
  int t = (idx >> 8) % NTOKENS;
  int b = idx / (256 * NTOKENS);
  float val;
  if (t == 0) {
    val = cls[d];
  } else {
    int p = t - 1, gy = p >> 3, gx = p & 7;
    float acc = pb[d];
#pragma unroll
    for (int c = 0; c < 3; c++)
#pragma unroll
      for (int ky = 0; ky < 4; ky++)
#pragma unroll
        for (int kx = 0; kx < 4; kx++)
          acc += inp[(((size_t)b * 3 + c) * 32 + gy * 4 + ky) * 32 + gx * 4 + kx] *
                 pw[((d * 3 + c) * 4 + ky) * 4 + kx];
    val = acc;
  }
  x[idx] = val + pos[t * DMODEL + d];
}

// ---------------------------------------------------------------------------
// Weight prep: f32 -> f16, laid out [K][N] for WMMA B operand.
// ---------------------------------------------------------------------------
__global__ __launch_bounds__(256) void prep_weights_kernel(
    const float* __restrict__ attn_in_w, const float* __restrict__ attn_out_w,
    const float* __restrict__ exp_w1, const float* __restrict__ exp_w2,
    _Float16* __restrict__ Bqkv, _Float16* __restrict__ Bout,
    _Float16* __restrict__ W1h, _Float16* __restrict__ W2h) {
  const int Sq = 256 * 768, So = 256 * 256, S1 = 256 * 512, S2 = 512 * 256;
  const int per = Sq + So + S1 + S2;  // 524288
  int idx = blockIdx.x * 256 + threadIdx.x;
  int layer = idx / per;
  int r = idx % per;
  if (r < Sq) {
    int k = r / 768, n = r % 768;
    Bqkv[(size_t)layer * Sq + r] = (_Float16)attn_in_w[(size_t)layer * Sq + (size_t)n * 256 + k];
  } else if ((r -= Sq) < So) {
    int k = r / 256, n = r % 256;
    Bout[(size_t)layer * So + r] = (_Float16)attn_out_w[(size_t)layer * So + (size_t)n * 256 + k];
  } else if ((r -= So) < S1) {
    W1h[(size_t)layer * S1 + r] = (_Float16)exp_w1[(size_t)layer * 4 * S1 + r];  // expert 0
  } else {
    r -= S1;
    W2h[(size_t)layer * S2 + r] = (_Float16)exp_w2[(size_t)layer * 4 * S2 + r];  // expert 0
  }
}

// ---------------------------------------------------------------------------
// LayerNorm (f32 in) -> f16 out. One wave per row of 256.
// ---------------------------------------------------------------------------
__global__ __launch_bounds__(256) void ln_f16_kernel(
    const float* __restrict__ x, const float* __restrict__ w,
    const float* __restrict__ b, _Float16* __restrict__ out, int rows) {
  int lane = threadIdx.x & 31;
  int row = blockIdx.x * 8 + (threadIdx.x >> 5);
  if (row >= rows) return;
  const float* xr = x + (size_t)row * DMODEL;
  float v[8];
  float s = 0.f;
#pragma unroll
  for (int j = 0; j < 8; j++) { v[j] = xr[lane + 32 * j]; s += v[j]; }
  float mu = wave_sum(s) * (1.0f / DMODEL);
  float vs = 0.f;
#pragma unroll
  for (int j = 0; j < 8; j++) { float d = v[j] - mu; vs += d * d; }
  float var = wave_sum(vs) * (1.0f / DMODEL);
  float rstd = rsqrtf(var + 1e-5f);
#pragma unroll
  for (int j = 0; j < 8; j++) {
    int d = lane + 32 * j;
    out[(size_t)row * DMODEL + d] = (_Float16)((v[j] - mu) * rstd * w[d] + b[d]);
  }
}

// ---------------------------------------------------------------------------
// WMMA GEMM: C[M,N] = act(A[M,K] @ B[K,N] + bias).  A,B f16; accumulate f32.
// One 16(M)x64(N) strip per wave: A fragment reused across 4 chained WMMAs
// per K-step; per-lane B traffic is 64 scalar f16 loads (16 k rows x 4 columns).
// K and N are compile-time so all per-K-step address math folds into the
// 24-bit instruction offsets of the global loads (no VALU in the hot loop).
// MODE 0: store f16.  MODE 1: exact GELU, store f16.  MODE 2: f32 += (residual).
// Fragment layouts per CDNA5 ISA:
//   A (16x32 f16): lane l -> m=l&15, kg=l>>4; halves = A[m][k0+kg*8..+7],
//                  A[m][k0+16+kg*8..+7]
//   B (32x16 f16): lane l -> n=n0+(l&15), kg=l>>4; halves = B[k0+kg*8..+7][n], B[k0+16+kg*8..+7][n]
//   C/D (16x16 f32): lane l -> n=l&15, vgpr i -> m=8*(l>>4)+i
// ---------------------------------------------------------------------------
template <int N, int MODE>
__device__ __forceinline__ void gemm_epilogue(
    v8f c, int cm0, int cn, const float* __restrict__ bias,
    float* __restrict__ outf, _Float16* __restrict__ outh) {
  float bv = bias[cn];
#pragma unroll
  for (int i = 0; i < 8; i++) {
    int cm = cm0 + i;
    float v = c[i] + bv;
    if (MODE == 1) v = 0.5f * v * (1.0f + erff(v * 0.70710678118654752f));
    if (MODE == 2) {
      outf[(size_t)cm * N + cn] += v;                 // residual add into x (f32)
    } else {
      outh[(size_t)cm * N + cn] = (_Float16)v;
    }
  }
}

template <int K, int N, int MODE>
__global__ __launch_bounds__(256) void wmma_gemm_kernel(
    const _Float16* __restrict__ A, const _Float16* __restrict__ Bm,
    const float* __restrict__ bias, float* __restrict__ outf,
    _Float16* __restrict__ outh) {
  int lane = threadIdx.x & 31;
  int t = blockIdx.x * 8 + (threadIdx.x >> 5);
  constexpr int stripsN = N / 64;       // 64-wide N strips
  int tm = t / stripsN, tn = t % stripsN;
  int ml = lane & 15, kg = lane >> 4;
  const _Float16* arow = A + (size_t)(tm * 16 + ml) * K;
  const _Float16* brow = Bm + (size_t)(kg * 8) * N + tn * 64 + ml;  // k-advance is constant
  v8f c0 = {}, c1 = {}, c2 = {}, c3 = {};
#pragma unroll
  for (int k0 = 0; k0 < K; k0 += 32) {
    HF16 a;
    a.h[0] = *(const v8h*)(arow + k0 + kg * 8);
    a.h[1] = *(const v8h*)(arow + k0 + 16 + kg * 8);
    const _Float16* bp = brow + (size_t)k0 * N;  // compile-time fold per step
    HF16 b0, b1, b2, b3;
#pragma unroll
    for (int s = 0; s < 8; s++) {
      b0.v[s] = bp[(size_t)s * N +  0]; b0.v[8 + s] = bp[(size_t)(16 + s) * N +  0];
      b1.v[s] = bp[(size_t)s * N + 16]; b1.v[8 + s] = bp[(size_t)(16 + s) * N + 16];
      b2.v[s] = bp[(size_t)s * N + 32]; b2.v[8 + s] = bp[(size_t)(16 + s) * N + 32];
      b3.v[s] = bp[(size_t)s * N + 48]; b3.v[8 + s] = bp[(size_t)(16 + s) * N + 48];
    }
    c0 = __builtin_amdgcn_wmma_f32_16x16x32_f16(false, a.v, false, b0.v, (short)0, c0, false, false);
    c1 = __builtin_amdgcn_wmma_f32_16x16x32_f16(false, a.v, false, b1.v, (short)0, c1, false, false);
    c2 = __builtin_amdgcn_wmma_f32_16x16x32_f16(false, a.v, false, b2.v, (short)0, c2, false, false);
    c3 = __builtin_amdgcn_wmma_f32_16x16x32_f16(false, a.v, false, b3.v, (short)0, c3, false, false);
  }
  int cm0 = tm * 16 + (lane >> 4) * 8;
  int cn  = tn * 64 + (lane & 15);
  gemm_epilogue<N, MODE>(c0, cm0, cn +  0, bias, outf, outh);
  gemm_epilogue<N, MODE>(c1, cm0, cn + 16, bias, outf, outh);
  gemm_epilogue<N, MODE>(c2, cm0, cn + 32, bias, outf, outh);
  gemm_epilogue<N, MODE>(c3, cm0, cn + 48, bias, outf, outh);
}

// ---------------------------------------------------------------------------
// Attention: one wave per (batch, head, query token). 65 keys, head dim 64.
// qkv f16 [M, 768] (q|k|v). Writes o f16 [M, 256].
// ---------------------------------------------------------------------------
__global__ __launch_bounds__(256) void attn_kernel(
    const _Float16* __restrict__ qkv, _Float16* __restrict__ o) {
  __shared__ float alf[8][72];
  int lane = threadIdx.x & 31;
  int w = threadIdx.x >> 5;
  int wid = blockIdx.x * 8 + w;
  int b = wid / (NHEADS * NTOKENS);
  int r = wid % (NHEADS * NTOKENS);
  int hh = r / NTOKENS;
  int qt = r % NTOKENS;
  const _Float16* qrow = qkv + (size_t)(b * NTOKENS + qt) * 768 + hh * 64;
  float s[3];
#pragma unroll
  for (int j = 0; j < 3; j++) {
    int kt = lane + 32 * j;
    s[j] = -1e30f;
    if (kt < NTOKENS) {
      const _Float16* krow = qkv + (size_t)(b * NTOKENS + kt) * 768 + 256 + hh * 64;
      float acc = 0.f;
      for (int d = 0; d < 64; d++) acc += (float)qrow[d] * (float)krow[d];
      s[j] = acc * 0.125f;  // 1/sqrt(64)
    }
  }
  float m = wave_max(fmaxf(fmaxf(s[0], s[1]), s[2]));
  float p[3];
  float lsum = 0.f;
#pragma unroll
  for (int j = 0; j < 3; j++) {
    int kt = lane + 32 * j;
    p[j] = (kt < NTOKENS) ? expf(s[j] - m) : 0.f;
    lsum += p[j];
  }
  lsum = wave_sum(lsum);
  float inv = 1.0f / lsum;
#pragma unroll
  for (int j = 0; j < 3; j++) {
    int kt = lane + 32 * j;
    if (kt < NTOKENS) alf[w][kt] = p[j] * inv;
  }
  __syncthreads();
#pragma unroll
  for (int j = 0; j < 2; j++) {
    int d = lane + 32 * j;
    float acc = 0.f;
    for (int kt = 0; kt < NTOKENS; kt++)
      acc += alf[w][kt] * (float)qkv[(size_t)(b * NTOKENS + kt) * 768 + 512 + hh * 64 + d];
    o[(size_t)(b * NTOKENS + qt) * 256 + hh * 64 + d] = (_Float16)acc;
  }
}

// ---------------------------------------------------------------------------
// Classification head on cls token: optional LayerNorm, then x @ W^T + b.
// One block (256 threads) per batch element.
// ---------------------------------------------------------------------------
__global__ __launch_bounds__(256) void head_kernel(
    const float* __restrict__ x, const float* __restrict__ w,
    const float* __restrict__ bias, const float* __restrict__ lnw,
    const float* __restrict__ lnb, float* __restrict__ out, int do_ln) {
  __shared__ float row[DMODEL];
  __shared__ float red[DMODEL];
  int b = blockIdx.x;
  int tid = threadIdx.x;
  float v = x[(size_t)b * NTOKENS * DMODEL + tid];
  if (do_ln) {
    red[tid] = v;
    __syncthreads();
    for (int s2 = 128; s2 > 0; s2 >>= 1) {
      if (tid < s2) red[tid] += red[tid + s2];
      __syncthreads();
    }
    float mu = red[0] * (1.0f / DMODEL);
    __syncthreads();
    float dv = v - mu;
    red[tid] = dv * dv;
    __syncthreads();
    for (int s2 = 128; s2 > 0; s2 >>= 1) {
      if (tid < s2) red[tid] += red[tid + s2];
      __syncthreads();
    }
    float var = red[0] * (1.0f / DMODEL);
    float rstd = rsqrtf(var + 1e-5f);
    v = dv * rstd * lnw[tid] + lnb[tid];
  }
  row[tid] = v;
  __syncthreads();
  if (tid < NCLASSES) {
    float acc = bias[tid];
    for (int d = 0; d < DMODEL; d++) acc += row[d] * w[tid * DMODEL + d];
    out[(size_t)b * NCLASSES + tid] = acc;
  }
}

// ---------------------------------------------------------------------------
extern "C" void kernel_launch(void* const* d_in, const int* in_sizes, int n_in,
                              void* d_out, int out_size, void* d_ws, size_t ws_size,
                              hipStream_t stream) {
  (void)in_sizes; (void)n_in; (void)out_size; (void)ws_size;
  const float* x_in    = (const float*)d_in[0];
  const float* patch_w = (const float*)d_in[1];
  const float* patch_b = (const float*)d_in[2];
  const float* cls     = (const float*)d_in[3];
  const float* pos     = (const float*)d_in[4];
  const float* ln1_w   = (const float*)d_in[5];
  const float* ln1_b   = (const float*)d_in[6];
  const float* ain_w   = (const float*)d_in[7];
  const float* ain_b   = (const float*)d_in[8];
  const float* aout_w  = (const float*)d_in[9];
  const float* aout_b  = (const float*)d_in[10];
  const float* ln2_w   = (const float*)d_in[11];
  const float* ln2_b   = (const float*)d_in[12];
  // d_in[13..18] (GAT/gnorm) and d_in[29] (edge_index) are mathematically dead:
  // the gate softmax is over a singleton axis, so the MoE weight is exactly 1.
  const float* ew1     = (const float*)d_in[19];
  const float* eb1     = (const float*)d_in[20];
  const float* ew2     = (const float*)d_in[21];
  const float* eb2     = (const float*)d_in[22];
  const float* norm_w  = (const float*)d_in[23];
  const float* norm_b  = (const float*)d_in[24];
  const float* head_w  = (const float*)d_in[25];
  const float* head_b  = (const float*)d_in[26];
  const float* aux_w   = (const float*)d_in[27];
  const float* aux_b   = (const float*)d_in[28];

  char* ws = (char*)d_ws;
  size_t off = 0;
  auto alloc = [&](size_t bytes) -> char* {
    char* p = ws + off;
    off += (bytes + 255) & ~(size_t)255;
    return p;
  };
  float*    x     = (float*)   alloc((size_t)MROWS * DMODEL * 4);   // 34.1 MB
  _Float16* h16   = (_Float16*)alloc((size_t)MROWS * DMODEL * 2);   // 17.0 MB
  _Float16* qkv16 = (_Float16*)alloc((size_t)MROWS * 768 * 2);      // 51.1 MB
  _Float16* o16   = (_Float16*)alloc((size_t)MROWS * DMODEL * 2);   // 17.0 MB
  _Float16* mid16 = (_Float16*)alloc((size_t)MROWS * 512 * 2);      // 34.1 MB
  _Float16* Bqkv  = (_Float16*)alloc((size_t)LAYERS * 256 * 768 * 2);
  _Float16* Bout  = (_Float16*)alloc((size_t)LAYERS * 256 * 256 * 2);
  _Float16* W1h   = (_Float16*)alloc((size_t)LAYERS * 256 * 512 * 2);
  _Float16* W2h   = (_Float16*)alloc((size_t)LAYERS * 512 * 256 * 2);

  float* out = (float*)d_out;  // [512,100] head ++ [512,100] aux

  prep_weights_kernel<<<(LAYERS * 524288) / 256, 256, 0, stream>>>(
      ain_w, aout_w, ew1, ew2, Bqkv, Bout, W1h, W2h);
  patch_embed_kernel<<<(BATCH * NTOKENS * DMODEL) / 256, 256, 0, stream>>>(
      x_in, patch_w, patch_b, cls, pos, x);

  const int MT = MROWS / 16;  // 2080 M-tiles; waves cover 16x64 strips
  for (int i = 0; i < LAYERS; i++) {
    ln_f16_kernel<<<MROWS / 8, 256, 0, stream>>>(
        x, ln1_w + i * DMODEL, ln1_b + i * DMODEL, h16, MROWS);
    wmma_gemm_kernel<256, 768, 0><<<MT * (768 / 64) / 8, 256, 0, stream>>>(
        h16, Bqkv + (size_t)i * 256 * 768, ain_b + i * 768, nullptr, qkv16);
    attn_kernel<<<(BATCH * NHEADS * NTOKENS) / 8, 256, 0, stream>>>(qkv16, o16);
    wmma_gemm_kernel<256, 256, 2><<<MT * (256 / 64) / 8, 256, 0, stream>>>(
        o16, Bout + (size_t)i * 256 * 256, aout_b + i * 256, x, nullptr);
    ln_f16_kernel<<<MROWS / 8, 256, 0, stream>>>(
        x, ln2_w + i * DMODEL, ln2_b + i * DMODEL, h16, MROWS);
    wmma_gemm_kernel<256, 512, 1><<<MT * (512 / 64) / 8, 256, 0, stream>>>(
        h16, W1h + (size_t)i * 256 * 512, eb1 + (size_t)i * 4 * 512, nullptr, mid16);
    wmma_gemm_kernel<512, 256, 2><<<MT * (256 / 64) / 8, 256, 0, stream>>>(
        mid16, W2h + (size_t)i * 512 * 256, eb2 + (size_t)i * 4 * 256, x, nullptr);
    if (i == 3)
      head_kernel<<<BATCH, 256, 0, stream>>>(
          x, aux_w, aux_b, nullptr, nullptr, out + (size_t)BATCH * NCLASSES, 0);
  }
  head_kernel<<<BATCH, 256, 0, stream>>>(x, head_w, head_b, norm_w, norm_b, out, 1);
}
